// illumination_module_73985106641404
// MI455X (gfx1250) — matrix operand order repaired and checked by reading, not compiled, not run on hardware
//
#include <hip/hip_runtime.h>
#include <hip/hip_bf16.h>
#include <math.h>

typedef __attribute__((ext_vector_type(16))) _Float16 v16h;
typedef __attribute__((ext_vector_type(8)))  _Float16 v8h;
typedef __attribute__((ext_vector_type(2)))  _Float16 v2h;
typedef __attribute__((ext_vector_type(8)))  float    v8f;

constexpr int B_  = 2;
constexpr int C_  = 64;
constexpr int HW_ = 96 * 96;   // 9216
constexpr int LP_ = 72;        // padded LDS row stride (f16) to spread banks
constexpr int NTILE_ = HW_ / 64;  // 144

#define DEV static __device__ __forceinline__

DEV v8h ld8h(const _Float16* p) { return *reinterpret_cast<const v8h*>(p); }

DEV v16h cat16(v8h a, v8h b) {
  return __builtin_shufflevector(a, b, 0,1,2,3,4,5,6,7,8,9,10,11,12,13,14,15);
}

// A-matrix (16x32 f16) fragment from row-major source: lane = row M,
// regs: VGPR0-3 = K(frag)=8h..8h+7, VGPR4-7 = K(frag)=16+8h..23+8h
DEV v16h frag_a(const _Float16* rowp, int k0, int h) {
  return cat16(ld8h(rowp + k0 + 8 * h), ld8h(rowp + k0 + 16 + 8 * h));
}

// B-matrix (32x16 f16) fragment: lane = N column, p points at (k0, n), row stride ld:
// elements K=8h..8h+7, then K=16+8h..23+8h (same order as frag_a)
DEV v16h frag_b(const _Float16* p, int ld, int h) {
  v16h r;
#pragma unroll
  for (int s = 0; s < 16; ++s) {
    const int k = (s < 8) ? (8 * h + s) : (16 + 8 * h + (s - 8));
    r[s] = p[k * ld];
  }
  return r;
}

DEV v8f wmma_f16(v16h a, v16h b, v8f c) {
  return __builtin_amdgcn_wmma_f32_16x16x32_f16(false, a, false, b, (short)0, c,
                                                false, false);
}

// CDNA5 async copy: global -> LDS, 16B per lane, tracked by ASYNCcnt.
// VDST = per-lane LDS byte offset; VADDR = per-lane 64-bit global address.
DEV void async_ld_b128(_Float16* ldst, const _Float16* gsrc) {
  unsigned loff = (unsigned)(size_t)ldst;   // AS(3) offset = low 32 bits
  asm volatile("global_load_async_to_lds_b128 %0, %1, off"
               :: "v"(loff), "v"(gsrc) : "memory");
}
DEV void wait_async0() { asm volatile("s_wait_asynccnt 0x0" ::: "memory"); }

// ---------------------------------------------------------------------------
// Stage 1: e = w_sp @ x + b_sp ; emit f16 in both [c][hw] and [hw][c] layouts.
// One block computes all 64 channels for a 128-column slab (x read once).
// ---------------------------------------------------------------------------
__global__ __launch_bounds__(128) void k_compute_e(const float* __restrict__ x,
                                                   const float* __restrict__ w_sp,
                                                   const float* __restrict__ b_sp,
                                                   _Float16* __restrict__ e16,
                                                   _Float16* __restrict__ et16) {
  __shared__ float xs[64 * 128];   // x tile  [k][n]
  __shared__ float ws[64 * 64];    // w_sp    [c][k]
  const int tid = threadIdx.x;
  const int n0  = blockIdx.x * 128;
  const int b   = blockIdx.y;
  const size_t bo = (size_t)b * C_ * HW_;

#pragma unroll
  for (int i = 0; i < 8; ++i)
    ((float4*)ws)[i * 128 + tid] = ((const float4*)w_sp)[i * 128 + tid];
#pragma unroll
  for (int i = 0; i < 16; ++i) {
    const int idx4 = i * 128 + tid;
    const int row  = idx4 >> 5;
    const int col  = (idx4 & 31) * 4;
    *(float4*)&xs[row * 128 + col] =
        *(const float4*)&x[bo + (size_t)row * HW_ + n0 + col];
  }
  __syncthreads();

  const int n = tid;
  _Float16 etbuf[64];
#pragma unroll 4
  for (int c = 0; c < C_; ++c) {
    float acc = b_sp[c];
#pragma unroll 8
    for (int k = 0; k < C_; ++k) acc += ws[c * 64 + k] * xs[k * 128 + n];
    const _Float16 v = (_Float16)acc;
    e16[bo + (size_t)c * HW_ + n0 + n] = v;
    etbuf[c] = v;
  }
#pragma unroll
  for (int i = 0; i < 8; ++i)
    *(v8h*)&et16[bo + (size_t)(n0 + n) * C_ + i * 8] = *(v8h*)&etbuf[i * 8];
}

// ---------------------------------------------------------------------------
// Stage 2a: per-row softmax stats M_n, L_n over S = E^T E
// (WMMA; E tile double-buffered in LDS via async-to-LDS DMA)
// ---------------------------------------------------------------------------
__global__ __launch_bounds__(128) void k_attn_stats(const _Float16* __restrict__ e16,
                                                    const _Float16* __restrict__ et16,
                                                    float* __restrict__ Mn,
                                                    float* __restrict__ Ln) {
  __shared__ _Float16 ldsE[2][64 * LP_];   // double-buffered 64x64 E tile

  const int b    = blockIdx.y;
  const int tid  = threadIdx.x;
  const int wv   = tid >> 5;
  const int lane = tid & 31;
  const int h    = lane >> 4;
  const int l16  = lane & 15;
  const int n0   = blockIdx.x * 64 + wv * 16;   // this wave's 16 query rows

  const _Float16* e  = e16  + (size_t)b * C_ * HW_;  // [64][HW]
  const _Float16* et = et16 + (size_t)b * C_ * HW_;  // [HW][64]

  const _Float16* qrow = et + (size_t)(n0 + l16) * C_;
  const v16h aq0 = frag_a(qrow, 0, h);
  const v16h aq1 = frag_a(qrow, 32, h);

  // cooperative staging: 8 lanes per 128B row -> 4 x b128 per thread
  const int srow0 = tid >> 3;          // base row, +16 per segment
  const int soff  = (tid & 7) * 8;     // f16 column offset within row

  float mrun[8], lrun[8];
#pragma unroll
  for (int r = 0; r < 8; ++r) { mrun[r] = -1e30f; lrun[r] = 0.f; }

  // preload tile 0 (async DMA to LDS)
#pragma unroll
  for (int it = 0; it < 4; ++it) {
    const int row = srow0 + it * 16;
    async_ld_b128(&ldsE[0][row * LP_ + soff], e + (size_t)row * HW_ + soff);
  }
  wait_async0();
  __syncthreads();

  const int dlo = l16 + 16 * h;        // B-frag K row, chunk0
  for (int t = 0; t < NTILE_; ++t) {
    const int cur = t & 1, nxt = cur ^ 1;
    const bool have = (t + 1 < NTILE_);
    if (have) {
      const int m1 = (t + 1) * 64;
      __builtin_prefetch(e + (size_t)srow0 * HW_ + m1 + 64, 0, 1);
#pragma unroll
      for (int it = 0; it < 4; ++it) {
        const int row = srow0 + it * 16;
        async_ld_b128(&ldsE[nxt][row * LP_ + soff],
                      e + (size_t)row * HW_ + m1 + soff);
      }
    }

    v8f s[4];
#pragma unroll
    for (int mt = 0; mt < 4; ++mt) {
      const v16h b0 = frag_b(&ldsE[cur][mt * 16 + l16], LP_, h);
      const v16h b1 = frag_b(&ldsE[cur][32 * LP_ + mt * 16 + l16], LP_, h);
      v8f z = {0.f, 0.f, 0.f, 0.f, 0.f, 0.f, 0.f, 0.f};
      s[mt] = wmma_f16(aq1, b1, wmma_f16(aq0, b0, z));
    }
#pragma unroll
    for (int r = 0; r < 8; ++r) {
      float tmax = fmaxf(fmaxf(s[0][r], s[1][r]), fmaxf(s[2][r], s[3][r]));
      tmax = fmaxf(tmax, __shfl_xor(tmax, 1));
      tmax = fmaxf(tmax, __shfl_xor(tmax, 2));
      tmax = fmaxf(tmax, __shfl_xor(tmax, 4));
      tmax = fmaxf(tmax, __shfl_xor(tmax, 8));
      const float mnew = fmaxf(mrun[r], tmax);
      float ts = __expf(s[0][r] - mnew) + __expf(s[1][r] - mnew) +
                 __expf(s[2][r] - mnew) + __expf(s[3][r] - mnew);
      ts += __shfl_xor(ts, 1);
      ts += __shfl_xor(ts, 2);
      ts += __shfl_xor(ts, 4);
      ts += __shfl_xor(ts, 8);
      lrun[r] = lrun[r] * __expf(mrun[r] - mnew) + ts;
      mrun[r] = mnew;
    }

    if (have) wait_async0();   // drain own DMA before signaling barrier
    __syncthreads();
  }
#pragma unroll
  for (int r = 0; r < 8; ++r) {
    if (l16 == r) {
      const int gn = n0 + r + 8 * h;
      Mn[(size_t)b * HW_ + gn] = mrun[r];
      Ln[(size_t)b * HW_ + gn] = lrun[r];
    }
  }
}

// ---------------------------------------------------------------------------
// Stage 2b: x1[c,m] = sum_n E[c,n] * exp(S[m,n]-M_n)/L_n  + x
// Both WMMA operand streams come from one async-staged LDS E tile.
// ---------------------------------------------------------------------------
__global__ __launch_bounds__(128) void k_attn_out(const _Float16* __restrict__ e16,
                                                  const _Float16* __restrict__ et16,
                                                  const float* __restrict__ Mn,
                                                  const float* __restrict__ Ln,
                                                  const float* __restrict__ x,
                                                  float* __restrict__ x1) {
  __shared__ _Float16 ldsE[2][64 * LP_];   // double-buffered 64x64 E tile
  __shared__ _Float16 ldsW[4][64 * 16];    // per-wave W^T tile: [n_local][m_local]

  const int b    = blockIdx.y;
  const int tid  = threadIdx.x;
  const int wv   = tid >> 5;
  const int lane = tid & 31;
  const int h    = lane >> 4;
  const int l16  = lane & 15;
  const int m0w  = blockIdx.x * 64 + wv * 16;  // this wave's 16 output columns m

  const _Float16* e  = e16  + (size_t)b * C_ * HW_;
  const _Float16* et = et16 + (size_t)b * C_ * HW_;
  const float* mnp = Mn + (size_t)b * HW_;
  const float* lnp = Ln + (size_t)b * HW_;

  const _Float16* qrow = et + (size_t)(m0w + l16) * C_;
  const v16h aq0 = frag_a(qrow, 0, h);
  const v16h aq1 = frag_a(qrow, 32, h);

  v8f o[4];
#pragma unroll
  for (int ct = 0; ct < 4; ++ct) o[ct] = (v8f){0.f,0.f,0.f,0.f,0.f,0.f,0.f,0.f};

  const int srow0 = tid >> 3;
  const int soff  = (tid & 7) * 8;
  const int dlo   = l16 + 16 * h;
  _Float16* wrow = &ldsW[wv][0];

  // preload tile 0 (async DMA to LDS)
#pragma unroll
  for (int it = 0; it < 4; ++it) {
    const int row = srow0 + it * 16;
    async_ld_b128(&ldsE[0][row * LP_ + soff], e + (size_t)row * HW_ + soff);
  }
  wait_async0();
  __syncthreads();

  for (int t = 0; t < NTILE_; ++t) {
    const int n0  = t * 64;
    const int cur = t & 1, nxt = cur ^ 1;
    const bool have = (t + 1 < NTILE_);
    if (have) {
      const int n1 = n0 + 64;
      __builtin_prefetch(e + (size_t)srow0 * HW_ + n1 + 64, 0, 1);
#pragma unroll
      for (int it = 0; it < 4; ++it) {
        const int row = srow0 + it * 16;
        async_ld_b128(&ldsE[nxt][row * LP_ + soff],
                      e + (size_t)row * HW_ + n1 + soff);
      }
    }

    // ---- S[m(16), n(64)] tiles, weighted -> LDS as W^T[n][m] (f16) ----
#pragma unroll
    for (int nt = 0; nt < 4; ++nt) {
      const v16h b0 = frag_b(&ldsE[cur][nt * 16 + l16], LP_, h);
      const v16h b1 = frag_b(&ldsE[cur][32 * LP_ + nt * 16 + l16], LP_, h);
      v8f z = {0.f, 0.f, 0.f, 0.f, 0.f, 0.f, 0.f, 0.f};
      v8f s = wmma_f16(aq1, b1, wmma_f16(aq0, b0, z));
      const int nn = n0 + nt * 16;
      const float mn  = mnp[nn + l16];          // per-lane (n) stats
      const float rln = 1.0f / lnp[nn + l16];
#pragma unroll
      for (int r = 0; r < 8; r += 2) {
        const float w0 = __expf(s[r] - mn) * rln;
        const float w1 = __expf(s[r + 1] - mn) * rln;
        v2h pk = {(_Float16)w0, (_Float16)w1};
        *(v2h*)&wrow[(nt * 16 + l16) * 16 + (r + 8 * h)] = pk;
      }
    }
    // ---- x1 tile accumulation: A = E rows (M=c) from LDS, B = W^T from LDS --
    const v16h bw0 = frag_b(wrow + l16, 16, h);
    const v16h bw1 = frag_b(wrow + 32 * 16 + l16, 16, h);
#pragma unroll
    for (int ct = 0; ct < 4; ++ct) {
      const _Float16* crow = &ldsE[cur][(ct * 16 + l16) * LP_];
      const v16h aE0 = frag_a(crow, 0, h);
      const v16h aE1 = frag_a(crow, 32, h);
      o[ct] = wmma_f16(aE1, bw1, wmma_f16(aE0, bw0, o[ct]));
    }

    if (have) wait_async0();   // drain own DMA before signaling barrier
    __syncthreads();
  }
  // ---- store x1 = O + x (residual) ----
  const size_t bo = (size_t)b * C_ * HW_;
#pragma unroll
  for (int ct = 0; ct < 4; ++ct) {
#pragma unroll
    for (int r = 0; r < 8; ++r) {
      const int c = ct * 16 + r + 8 * h;
      const size_t idx = bo + (size_t)c * HW_ + (m0w + l16);
      x1[idx] = o[ct][r] + x[idx];
    }
  }
}

// ---------------------------------------------------------------------------
// SE branch: y = mean(x, hw) ; wei = sigmoid(prelu(prelu(y W_c^T+b)W_r^T+b))
// ---------------------------------------------------------------------------
__global__ __launch_bounds__(256) void k_se_reduce(const float* __restrict__ x,
                                                   float* __restrict__ y) {
  __shared__ float red[256];
  const int bc = blockIdx.x;
  const float* p = x + (size_t)bc * HW_;
  float s = 0.f;
  for (int i = threadIdx.x; i < HW_; i += 256) s += p[i];
  red[threadIdx.x] = s;
  __syncthreads();
  for (int st = 128; st > 0; st >>= 1) {
    if (threadIdx.x < st) red[threadIdx.x] += red[threadIdx.x + st];
    __syncthreads();
  }
  if (threadIdx.x == 0) y[bc] = red[0] * (1.0f / HW_);
}

__global__ __launch_bounds__(128) void k_se_mlp(const float* __restrict__ y,
                                                const float* __restrict__ w_comp,
                                                const float* __restrict__ b_comp,
                                                const float* __restrict__ a_comp,
                                                const float* __restrict__ w_rec,
                                                const float* __restrict__ b_rec,
                                                const float* __restrict__ a_rec,
                                                float* __restrict__ wei) {
  __shared__ float ys[B_ * C_], ps[B_ * C_];
  const int t = threadIdx.x;
  ys[t] = y[t];
  __syncthreads();
  const int b = t >> 6, o = t & 63;
  float acc = b_comp[o];
#pragma unroll 8
  for (int k = 0; k < C_; ++k) acc += ys[b * C_ + k] * w_comp[o * C_ + k];
  const float ac = a_comp[0];
  ps[t] = acc >= 0.f ? acc : ac * acc;
  __syncthreads();
  float acc2 = b_rec[o];
#pragma unroll 8
  for (int k = 0; k < C_; ++k) acc2 += ps[b * C_ + k] * w_rec[o * C_ + k];
  const float ar = a_rec[0];
  const float r = acc2 >= 0.f ? acc2 : ar * acc2;
  wei[t] = 1.0f / (1.0f + __expf(-r));
}

// ---------------------------------------------------------------------------
// Epilogue: out = prelu(W_out @ [x1 ; x*wei] + b_out) + x
// One block computes all 64 channels for a 128-column slab (x1/x read once).
// ---------------------------------------------------------------------------
__global__ __launch_bounds__(128) void k_epilogue(const float* __restrict__ x1,
                                                  const float* __restrict__ x,
                                                  const float* __restrict__ wei,
                                                  const float* __restrict__ w_out,
                                                  const float* __restrict__ b_out,
                                                  const float* __restrict__ a_out,
                                                  float* __restrict__ out) {
  __shared__ float s1[64 * 128];   // x1 tile [k][n]
  __shared__ float s2[64 * 128];   // x  tile [k][n]
  __shared__ float wsh[64 * 128];  // w_out [c][2C] (2nd half folded with wei)
  const int tid = threadIdx.x;
  const int n0  = blockIdx.x * 128;
  const int b   = blockIdx.y;
  const size_t bo = (size_t)b * C_ * HW_;

#pragma unroll
  for (int i = 0; i < 16; ++i)
    ((float4*)wsh)[i * 128 + tid] = ((const float4*)w_out)[i * 128 + tid];
#pragma unroll
  for (int i = 0; i < 16; ++i) {
    const int idx4 = i * 128 + tid;
    const int row  = idx4 >> 5;
    const int col  = (idx4 & 31) * 4;
    *(float4*)&s1[row * 128 + col] =
        *(const float4*)&x1[bo + (size_t)row * HW_ + n0 + col];
    *(float4*)&s2[row * 128 + col] =
        *(const float4*)&x[bo + (size_t)row * HW_ + n0 + col];
  }
  __syncthreads();
  // fold wei into second-half weights: wsh[c][64+k] *= wei[b][k]
#pragma unroll
  for (int i = 0; i < 32; ++i) {
    const int idx = i * 128 + tid;       // 4096 elements (c,k)
    const int c = idx >> 6, k = idx & 63;
    wsh[c * 128 + 64 + k] *= wei[b * C_ + k];
  }
  __syncthreads();

  const int n = tid;
  const float ao = a_out[0];
#pragma unroll 2
  for (int c = 0; c < C_; ++c) {
    float acc = b_out[c];
#pragma unroll 8
    for (int k = 0; k < C_; ++k)
      acc += wsh[c * 128 + k] * s1[k * 128 + n] +
             wsh[c * 128 + 64 + k] * s2[k * 128 + n];
    const float v = acc >= 0.f ? acc : ao * acc;
    out[bo + (size_t)c * HW_ + n0 + n] = v + s2[c * 128 + n];
  }
}

// ---------------------------------------------------------------------------
extern "C" void kernel_launch(void* const* d_in, const int* in_sizes, int n_in,
                              void* d_out, int out_size, void* d_ws, size_t ws_size,
                              hipStream_t stream) {
  (void)in_sizes; (void)n_in; (void)out_size; (void)ws_size;
  const float* x      = (const float*)d_in[0];
  const float* w_sp   = (const float*)d_in[1];
  const float* b_sp   = (const float*)d_in[2];
  const float* w_comp = (const float*)d_in[3];
  const float* b_comp = (const float*)d_in[4];
  const float* a_comp = (const float*)d_in[5];
  const float* w_rec  = (const float*)d_in[6];
  const float* b_rec  = (const float*)d_in[7];
  const float* a_rec  = (const float*)d_in[8];
  const float* w_out  = (const float*)d_in[9];
  const float* b_out  = (const float*)d_in[10];
  const float* a_out  = (const float*)d_in[11];
  float* out = (float*)d_out;

  // workspace carve-up
  const size_t nBCHW = (size_t)B_ * C_ * HW_;
  _Float16* e16  = (_Float16*)d_ws;
  _Float16* et16 = e16 + nBCHW;
  float* Mn  = (float*)(et16 + nBCHW);
  float* Ln  = Mn + (size_t)B_ * HW_;
  float* x1  = Ln + (size_t)B_ * HW_;
  float* y   = x1 + nBCHW;
  float* wei = y + B_ * C_;

  k_compute_e<<<dim3(HW_ / 128, B_), 128, 0, stream>>>(x, w_sp, b_sp, e16, et16);
  k_se_reduce<<<B_ * C_, 256, 0, stream>>>(x, y);
  k_se_mlp<<<1, 128, 0, stream>>>(y, w_comp, b_comp, a_comp, w_rec, b_rec, a_rec, wei);
  k_attn_stats<<<dim3(HW_ / 64, B_), 128, 0, stream>>>(e16, et16, Mn, Ln);
  k_attn_out<<<dim3(HW_ / 64, B_), 128, 0, stream>>>(e16, et16, Mn, Ln, x, x1);
  k_epilogue<<<dim3(HW_ / 128, B_), 128, 0, stream>>>(x1, x, wei, w_out, b_out, a_out, out);
}
